// PFNPredictorNodeCls_56521769616167
// MI455X (gfx1250) — compile-verified, matrix operand order checked by reading, not run on hardware
//
#include <hip/hip_runtime.h>
#include <hip/hip_bf16.h>

typedef unsigned short u16;
typedef unsigned int   u32;

typedef __attribute__((ext_vector_type(16))) __bf16 v16bf;
typedef __attribute__((ext_vector_type(8)))  float  v8f;

#define NUM_EXPERTS 8
#define MOE_TOP_K   2
#define N_TOKENS    4096
#define HIDDEN      1024
#define EXPERT_DIM  4096
#define SLOTS       (N_TOKENS * MOE_TOP_K)

// GEMM tiling: 128x128 tile, K-step 32 (one bf16 WMMA K), 256 threads = 8 waves,
// double-buffered LDS filled by async global->LDS B128 copies.
#define TM  128
#define TN  128
#define TK  32
#define LDH 40   // halfwords per LDS row (32 data + 8 pad); 80B stride = 16B aligned

__device__ __forceinline__ u16 f2bf(float x) {
  u32 u = __float_as_uint(x);
  u += 0x7FFFu + ((u >> 16) & 1u);   // round-to-nearest-even
  return (u16)(u >> 16);
}

// Truncate a generic pointer to a __shared__ object to its 32-bit LDS offset
// (flat LDS aperture keeps the offset in addr[31:0]).
__device__ __forceinline__ u32 lds_off(const void* p) {
  return (u32)(uintptr_t)p;
}

// CDNA5 async DMA: copy 16 bytes global -> LDS, tracked by ASYNCcnt.
__device__ __forceinline__ void async_copy_b128(u32 lds_addr, const u16* gaddr) {
  asm volatile("global_load_async_to_lds_b128 %0, %1, off"
               :: "v"(lds_addr), "v"(gaddr)
               : "memory");
}

// Load one 16x32 bf16 WMMA fragment from an LDS tile whose rows are K-contiguous.
// Matches ISA 7.12.2 16-bit A layout:
//   lanes 0-15 : V0..V3 = K 0..7,  V4..V7 = K 16..23 (row = lane)
//   lanes 16-31: V0..V3 = K 8..15, V4..V7 = K 24..31
__device__ __forceinline__ v16bf load_frag(const u16* tile, int row0, int lane) {
  const int r   = row0 + (lane & 15);
  const int kb2 = (lane >> 4) * 4;           // dword offset: 0 or 4
  const u32* p  = (const u32*)(tile + r * LDH);
  union { u32 u[8]; v16bf v; } f;
#pragma unroll
  for (int i = 0; i < 4; ++i) {
    f.u[i]     = p[kb2 + i];
    f.u[i + 4] = p[8 + kb2 + i];
  }
  return f.v;
}

// ---------------------------------------------------------------- gating ----
__global__ __launch_bounds__(256)
void moe_gate_kernel(const float* __restrict__ x, const float* __restrict__ Wg,
                     const float* __restrict__ bg, float* __restrict__ probs,
                     int* __restrict__ top_idx, float* __restrict__ top_gate,
                     int* __restrict__ counts) {
  const int lane = threadIdx.x & 31;
  const int wave = threadIdx.x >> 5;
  const int t = blockIdx.x * 8 + wave;       // one wave32 per token
  if (t >= N_TOKENS) return;

  float acc[NUM_EXPERTS];
#pragma unroll
  for (int e = 0; e < NUM_EXPERTS; ++e) acc[e] = 0.0f;
  const float* xr = x + (size_t)t * HIDDEN;
  for (int k = lane; k < HIDDEN; k += 32) {
    const float xv = xr[k];
    const float* wr = Wg + (size_t)k * NUM_EXPERTS;
#pragma unroll
    for (int e = 0; e < NUM_EXPERTS; ++e) acc[e] = fmaf(xv, wr[e], acc[e]);
  }
#pragma unroll
  for (int off = 16; off > 0; off >>= 1) {
#pragma unroll
    for (int e = 0; e < NUM_EXPERTS; ++e)
      acc[e] += __shfl_xor(acc[e], off, 32);
  }
  if (lane == 0) {
    float mx = -3.0e38f;
#pragma unroll
    for (int e = 0; e < NUM_EXPERTS; ++e) { acc[e] += bg[e]; mx = fmaxf(mx, acc[e]); }
    float s = 0.0f;
#pragma unroll
    for (int e = 0; e < NUM_EXPERTS; ++e) { acc[e] = __expf(acc[e] - mx); s += acc[e]; }
    const float inv = 1.0f / s;
#pragma unroll
    for (int e = 0; e < NUM_EXPERTS; ++e) {
      acc[e] *= inv;
      probs[(size_t)t * NUM_EXPERTS + e] = acc[e];
    }
    int i0 = 0; float g0 = acc[0];
#pragma unroll
    for (int e = 1; e < NUM_EXPERTS; ++e) if (acc[e] > g0) { g0 = acc[e]; i0 = e; }
    int i1 = -1; float g1 = -1.0f;
#pragma unroll
    for (int e = 0; e < NUM_EXPERTS; ++e)
      if (e != i0 && acc[e] > g1) { g1 = acc[e]; i1 = e; }
    const float gs = 1.0f / (g0 + g1);
    top_idx[t * 2 + 0] = i0;  top_gate[t * 2 + 0] = g0 * gs;
    top_idx[t * 2 + 1] = i1;  top_gate[t * 2 + 1] = g1 * gs;
    atomicAdd(&counts[i0], 1);
    atomicAdd(&counts[i1], 1);
  }
}

__global__ void moe_init_kernel(int* counts) {
  if (threadIdx.x < NUM_EXPERTS) counts[threadIdx.x] = 0;
}

__global__ void moe_offsets_kernel(const int* __restrict__ counts,
                                   int* __restrict__ offsets, int* __restrict__ fill) {
  if (threadIdx.x == 0) {
    int s = 0;
    for (int e = 0; e < NUM_EXPERTS; ++e) { offsets[e] = s; s += counts[e]; }
  }
  if (threadIdx.x < NUM_EXPERTS) fill[threadIdx.x] = 0;
}

__global__ __launch_bounds__(256)
void moe_scatter_kernel(const int* __restrict__ top_idx, const int* __restrict__ offsets,
                        int* __restrict__ fill, int* __restrict__ tok_of_slot,
                        int* __restrict__ slot_of_token) {
  const int t = blockIdx.x * 256 + threadIdx.x;
  if (t >= N_TOKENS) return;
#pragma unroll
  for (int k = 0; k < MOE_TOP_K; ++k) {
    const int e = top_idx[t * 2 + k];
    const int pos = atomicAdd(&fill[e], 1);
    const int s = offsets[e] + pos;
    tok_of_slot[s] = t;
    slot_of_token[t * 2 + k] = s;
  }
}

// ------------------------------------------------------------- conversion ---
__global__ __launch_bounds__(256)
void moe_cvt_bf16_kernel(const float* __restrict__ src, u16* __restrict__ dst, int n4) {
  const int i = blockIdx.x * 256 + threadIdx.x;
  if (i >= n4) return;
  const float4 f = ((const float4*)src)[i];
  uint2 o;
  o.x = (u32)f2bf(f.x) | ((u32)f2bf(f.y) << 16);
  o.y = (u32)f2bf(f.z) | ((u32)f2bf(f.w) << 16);
  ((uint2*)dst)[i] = o;
}

// Convert f32 [K][N] -> bf16 [N][K] (per expert), 64x64 tiles through LDS so the
// GEMM never has to transpose:  grid = (N/64, K/64, E).
template <int K, int N>
__global__ __launch_bounds__(256)
void moe_cvt_bf16_T_kernel(const float* __restrict__ src, u16* __restrict__ dst) {
  __shared__ float tile[64][65];               // 65-pad => conflict-free columns
  const int e  = blockIdx.z;
  const int k0 = blockIdx.y * 64;
  const int n0 = blockIdx.x * 64;
  const float* s = src + (size_t)e * K * N;
  u16* d = dst + (size_t)e * N * K;
  const int tid = threadIdx.x;
#pragma unroll
  for (int r = 0; r < 4; ++r) {                // coalesced f32 loads
    const int k = (tid >> 4) + r * 16;
    const int n = (tid & 15) * 4;
    const float4 f = *(const float4*)(s + (size_t)(k0 + k) * N + (n0 + n));
    tile[k][n + 0] = f.x; tile[k][n + 1] = f.y;
    tile[k][n + 2] = f.z; tile[k][n + 3] = f.w;
  }
  __syncthreads();
#pragma unroll
  for (int r = 0; r < 4; ++r) {                // coalesced bf16 stores, transposed
    const int n = (tid >> 4) + r * 16;
    const int k = (tid & 15) * 4;
    uint2 o;
    o.x = (u32)f2bf(tile[k + 0][n]) | ((u32)f2bf(tile[k + 1][n]) << 16);
    o.y = (u32)f2bf(tile[k + 2][n]) | ((u32)f2bf(tile[k + 3][n]) << 16);
    *(uint2*)(d + (size_t)(n0 + n) * K + (k0 + k)) = o;
  }
}

// ------------------------------------------------------------ WMMA GEMMs ----
// GATHER=true : GEMM1  h = relu(x[tok] @ W1[e] + b1[e]), bf16 out
// GATHER=false: GEMM2  y = h @ W2[e] + b2[e], f32 out
// A : [rows][KDIM] bf16 row-major.  BT : [E][NDIM][KDIM] bf16 (pre-transposed).
template <int KDIM, int NDIM, bool GATHER>
__global__ __launch_bounds__(256)
void moe_gemm_kernel(const u16* __restrict__ A, const u16* __restrict__ BT,
                     const float* __restrict__ bias,
                     const int* __restrict__ tok_of_slot,
                     const int* __restrict__ counts, const int* __restrict__ offsets,
                     u16* __restrict__ out_bf, float* __restrict__ out_f) {
  const int e = blockIdx.z;
  const int cnt = counts[e];
  const int m_base = blockIdx.y * TM;
  if (cnt == 0 || m_base >= cnt) return;       // early-exit empty tiles
  const int n_base = blockIdx.x * TN;
  const int sbase = offsets[e];

  __shared__ __align__(16) u16 As[2][TM][LDH]; // [buf][m][k]
  __shared__ __align__(16) u16 Bs[2][TN][LDH]; // [buf][n][k]

  const int tid  = threadIdx.x;
  const int lane = tid & 31;
  const int wave = tid >> 5;
  const int wm   = wave & 3;                   // 4 wave-rows  (32 rows each)
  const int wn   = wave >> 2;                  // 2 wave-cols  (64 cols each)

  // Tile loader: same mapping for A and B — row = tid>>1, halfword col 0 or 16;
  // each thread owns 32B of each tile per K-step => 2 async B128 copies each.
  const int lrow = tid >> 1;
  const int lcol = (tid & 1) * 16;
  int sl = m_base + lrow;  if (sl > cnt - 1) sl = cnt - 1;
  const size_t arow_g = GATHER ? (size_t)tok_of_slot[sbase + sl] : (size_t)(sbase + sl);
  const u16* agp = A + arow_g * KDIM + lcol;
  const u16* bgp = BT + ((size_t)e * NDIM + (size_t)(n_base + lrow)) * KDIM + lcol;
  const u32 a_lds[2] = { lds_off(&As[0][lrow][lcol]), lds_off(&As[1][lrow][lcol]) };
  const u32 b_lds[2] = { lds_off(&Bs[0][lrow][lcol]), lds_off(&Bs[1][lrow][lcol]) };

  v8f acc[2][4];
#pragma unroll
  for (int i = 0; i < 2; ++i)
#pragma unroll
    for (int j = 0; j < 4; ++j)
#pragma unroll
      for (int r = 0; r < 8; ++r) acc[i][j][r] = 0.0f;

  // Prime stage 0.
  async_copy_b128(a_lds[0],      agp);
  async_copy_b128(a_lds[0] + 16, agp + 8);
  async_copy_b128(b_lds[0],      bgp);
  async_copy_b128(b_lds[0] + 16, bgp + 8);

  for (int k0 = 0, it = 0; k0 < KDIM; k0 += TK, ++it) {
    const int p = it & 1;
    if (k0 + TK < KDIM) {
      const int q = p ^ 1;
      const u16* an = agp + (k0 + TK);
      const u16* bn = bgp + (k0 + TK);
      async_copy_b128(a_lds[q],      an);
      async_copy_b128(a_lds[q] + 16, an + 8);
      async_copy_b128(b_lds[q],      bn);
      async_copy_b128(b_lds[q] + 16, bn + 8);
      // 4 newest copies (next stage) may stay in flight; current stage is done.
      asm volatile("s_wait_asynccnt 0x4" ::: "memory");
    } else {
      asm volatile("s_wait_asynccnt 0x0" ::: "memory");
    }
    if (k0 + 8 * TK < KDIM)                    // keep weight stream warm in L2
      __builtin_prefetch((const void*)(bgp + k0 + 8 * TK), 0, 1);
    __syncthreads();

    v16bf af[2], bf[4];
#pragma unroll
    for (int i = 0; i < 2; ++i)
      af[i] = load_frag(&As[p][0][0], wm * 32 + i * 16, lane);
#pragma unroll
    for (int j = 0; j < 4; ++j)
      bf[j] = load_frag(&Bs[p][0][0], wn * 64 + j * 16, lane);
#pragma unroll
    for (int i = 0; i < 2; ++i)
#pragma unroll
      for (int j = 0; j < 4; ++j)
        acc[i][j] = __builtin_amdgcn_wmma_f32_16x16x32_bf16(
            false, af[i], false, bf[j], (short)0, acc[i][j], false, false);
    __syncthreads();                           // buffer p free for stage it+2
  }

  // Epilogue. C layout (ISA 7.12.2): VGPR r holds M=r (lanes 0-15) / M=r+8 (16-31).
  const int row_hi = (lane >> 4) * 8;
  const int ncol   = lane & 15;
#pragma unroll
  for (int i = 0; i < 2; ++i) {
#pragma unroll
    for (int j = 0; j < 4; ++j) {
      const int n = n_base + wn * 64 + j * 16 + ncol;
      const float bv = bias[(size_t)e * NDIM + n];
#pragma unroll
      for (int r = 0; r < 8; ++r) {
        const int s = m_base + wm * 32 + i * 16 + row_hi + r;
        if (s < cnt) {
          float v = acc[i][j][r] + bv;
          if (GATHER) {
            v = fmaxf(v, 0.0f);
            out_bf[(size_t)(sbase + s) * NDIM + n] = f2bf(v);
          } else {
            out_f[(size_t)(sbase + s) * NDIM + n] = v;
          }
        }
      }
    }
  }
}

// ---------------------------------------------------------------- combine ---
__global__ __launch_bounds__(256)
void moe_combine_kernel(const float* __restrict__ yb, const int* __restrict__ slot_of_token,
                        const float* __restrict__ tgate, float* __restrict__ out) {
  const int idx = blockIdx.x * 256 + threadIdx.x;   // over N_TOKENS * HIDDEN/4
  const int t = idx / (HIDDEN / 4);
  const int c = (idx % (HIDDEN / 4)) * 4;
  const int s0 = slot_of_token[t * 2 + 0];
  const int s1 = slot_of_token[t * 2 + 1];
  const float g0 = tgate[t * 2 + 0];
  const float g1 = tgate[t * 2 + 1];
  const float4 y0 = *(const float4*)(yb + (size_t)s0 * HIDDEN + c);
  const float4 y1 = *(const float4*)(yb + (size_t)s1 * HIDDEN + c);
  float4 o;
  o.x = g0 * y0.x + g1 * y1.x;
  o.y = g0 * y0.y + g1 * y1.y;
  o.z = g0 * y0.z + g1 * y1.z;
  o.w = g0 * y0.w + g1 * y1.w;
  *(float4*)(out + (size_t)t * HIDDEN + c) = o;
}

// aux = E * sum_e (sum_t probs[t][e] / N)^2.
__global__ __launch_bounds__(256)
void moe_aux_kernel(const float* __restrict__ probs, float* __restrict__ out_aux) {
  __shared__ float red[256];
  const int tid = threadIdx.x;
  const int e = tid & 7;
  const int chunk = tid >> 3;                        // 32 chunks of 128 tokens
  float s = 0.0f;
  for (int t = chunk * 128; t < (chunk + 1) * 128; ++t)
    s += probs[(size_t)t * NUM_EXPERTS + e];
  red[tid] = s;
  __syncthreads();
  if (tid < NUM_EXPERTS) {
    float tot = 0.0f;
    for (int c = 0; c < 32; ++c) tot += red[c * 8 + tid];
    red[tid] = tot;
  }
  __syncthreads();
  if (tid == 0) {
    float aux = 0.0f;
    for (int ee = 0; ee < NUM_EXPERTS; ++ee) {
      const float f = red[ee] * (1.0f / (float)N_TOKENS);
      aux += f * f;
    }
    out_aux[0] = (float)NUM_EXPERTS * aux;
  }
}

// ------------------------------------------------------------------ launch --
extern "C" void kernel_launch(void* const* d_in, const int* in_sizes, int n_in,
                              void* d_out, int out_size, void* d_ws, size_t ws_size,
                              hipStream_t stream) {
  (void)in_sizes; (void)n_in; (void)out_size; (void)ws_size;
  const float* x  = (const float*)d_in[0];
  const float* Wg = (const float*)d_in[1];
  const float* bg = (const float*)d_in[2];
  const float* W1 = (const float*)d_in[3];
  const float* b1 = (const float*)d_in[4];
  const float* W2 = (const float*)d_in[5];
  const float* b2 = (const float*)d_in[6];
  float* out = (float*)d_out;

  unsigned char* p = (unsigned char*)d_ws;
  auto carve = [&](size_t bytes) {
    void* r = (void*)p;
    p += (bytes + 255) & ~(size_t)255;
    return r;
  };
  u16*   xb    = (u16*)carve((size_t)N_TOKENS * HIDDEN * 2);                     //   8 MB
  u16*   w1t   = (u16*)carve((size_t)NUM_EXPERTS * HIDDEN * EXPERT_DIM * 2);     //  64 MB [E][N][K]
  u16*   w2t   = (u16*)carve((size_t)NUM_EXPERTS * EXPERT_DIM * HIDDEN * 2);     //  64 MB [E][N][K]
  u16*   h     = (u16*)carve((size_t)SLOTS * EXPERT_DIM * 2);                    //  64 MB
  float* yb    = (float*)carve((size_t)SLOTS * HIDDEN * 4);                      //  32 MB
  float* probs = (float*)carve((size_t)N_TOKENS * NUM_EXPERTS * 4);
  float* tgate = (float*)carve((size_t)N_TOKENS * MOE_TOP_K * 4);
  int*   tidx  = (int*)carve((size_t)N_TOKENS * MOE_TOP_K * 4);
  int*   tok_of_slot   = (int*)carve((size_t)SLOTS * 4);
  int*   slot_of_token = (int*)carve((size_t)SLOTS * 4);
  int*   counts  = (int*)carve(64);
  int*   offsets = (int*)carve(64);
  int*   fill    = (int*)carve(64);

  moe_init_kernel<<<1, 32, 0, stream>>>(counts);

  const int n4x = N_TOKENS * HIDDEN / 4;
  moe_cvt_bf16_kernel<<<(n4x + 255) / 256, 256, 0, stream>>>(x, xb, n4x);
  moe_cvt_bf16_T_kernel<HIDDEN, EXPERT_DIM>
      <<<dim3(EXPERT_DIM / 64, HIDDEN / 64, NUM_EXPERTS), 256, 0, stream>>>(W1, w1t);
  moe_cvt_bf16_T_kernel<EXPERT_DIM, HIDDEN>
      <<<dim3(HIDDEN / 64, EXPERT_DIM / 64, NUM_EXPERTS), 256, 0, stream>>>(W2, w2t);

  moe_gate_kernel<<<N_TOKENS / 8, 256, 0, stream>>>(x, Wg, bg, probs, tidx, tgate, counts);
  moe_offsets_kernel<<<1, 32, 0, stream>>>(counts, offsets, fill);
  moe_scatter_kernel<<<N_TOKENS / 256, 256, 0, stream>>>(tidx, offsets, fill,
                                                         tok_of_slot, slot_of_token);

  moe_gemm_kernel<HIDDEN, EXPERT_DIM, true>
      <<<dim3(EXPERT_DIM / TN, N_TOKENS / TM, NUM_EXPERTS), 256, 0, stream>>>(
          xb, w1t, b1, tok_of_slot, counts, offsets, h, nullptr);

  moe_gemm_kernel<EXPERT_DIM, HIDDEN, false>
      <<<dim3(HIDDEN / TN, N_TOKENS / TM, NUM_EXPERTS), 256, 0, stream>>>(
          h, w2t, b2, tok_of_slot, counts, offsets, nullptr, yb);

  moe_combine_kernel<<<N_TOKENS * (HIDDEN / 4) / 256, 256, 0, stream>>>(
      yb, slot_of_token, tgate, out);
  moe_aux_kernel<<<1, 256, 0, stream>>>(probs, out + (size_t)N_TOKENS * HIDDEN);
}
